// MMoELayer_32057635897965
// MI455X (gfx1250) — compile-verified, matrix operand order checked by reading, not run on hardware
//
#include <hip/hip_runtime.h>
#include <hip/hip_bf16.h>

typedef __attribute__((ext_vector_type(16))) __bf16     v16bf;
typedef __attribute__((ext_vector_type(8)))  float      v8f;
typedef __attribute__((ext_vector_type(8)))  unsigned   u32x8;

// ---------------------------------------------------------------------------
// Constants for this problem
// ---------------------------------------------------------------------------
#define IN_F    1536
#define KT      48          // 1536 / 32 k-tiles
#define NEXP    4
#define HID     64
#define NTASK   37
#define NGCOL   148         // 37*4 gate columns
#define NGT     10          // ceil(148/16) n-tiles for gates
#define BROWS   16384

// packed-fragment region sizes (bf16 elements)
#define BG_ELEMS  (KT * NGT * 512)          // 245760
#define B1_ELEMS  (NEXP * KT * 4 * 512)     // 393216
#define B2_ELEMS  (NEXP * 2 * 4 * 512)      // 16384

// workspace byte offsets
#define OFF_BG      (9699328ull)            // after gates f32 (16384*148*4)
#define OFF_B1      (10190848ull)
#define OFF_B2      (10977280ull)

// ---------------------------------------------------------------------------
// Helpers
// ---------------------------------------------------------------------------

// Load A fragment (16x32 bf16) for this wave's 16-row tile directly from f32 x.
// Lane layout (ISA 7.12.2, 16-bit A 16x32): lanes 0-15 row=lane, K {0..7,16..23};
// lanes 16-31 row=lane-16, K {8..15,24..31}.  kbase already includes half*8.
__device__ __forceinline__ v16bf load_a_from_x(const float* __restrict__ xrow, int kbase) {
    const float4* p = (const float4*)(xrow + kbase);
    float4 a0 = p[0], a1 = p[1], a2 = p[4], a3 = p[5];
    v16bf a;
    a[0]  = (__bf16)a0.x; a[1]  = (__bf16)a0.y; a[2]  = (__bf16)a0.z; a[3]  = (__bf16)a0.w;
    a[4]  = (__bf16)a1.x; a[5]  = (__bf16)a1.y; a[6]  = (__bf16)a1.z; a[7]  = (__bf16)a1.w;
    a[8]  = (__bf16)a2.x; a[9]  = (__bf16)a2.y; a[10] = (__bf16)a2.z; a[11] = (__bf16)a2.w;
    a[12] = (__bf16)a3.x; a[13] = (__bf16)a3.y; a[14] = (__bf16)a3.z; a[15] = (__bf16)a3.w;
    return a;
}

// Load a pre-packed B fragment: 32 bytes per lane, contiguous.
__device__ __forceinline__ v16bf load_b_frag(const __bf16* __restrict__ base, int lane) {
    u32x8 u = *(const u32x8*)(base + lane * 16);
    return __builtin_bit_cast(v16bf, u);
}

__device__ __forceinline__ v8f wmma_bf16(v16bf a, v16bf b, v8f c) {
    return __builtin_amdgcn_wmma_f32_16x16x32_bf16(false, a, false, b, (short)0, c, false, false);
}

__device__ __forceinline__ v8f vzero8() {
    v8f z;
#pragma unroll
    for (int j = 0; j < 8; ++j) z[j] = 0.0f;
    return z;
}

// ---------------------------------------------------------------------------
// Kernel 0: pack W1 / W2 / Wg (f32) into bf16 WMMA B-fragment layout.
// One thread per output element.  B-fragment lane layout mirrors the A layout:
// lane<16 -> col=lane, K {0..7,16..23}; lane>=16 -> col=lane-16, K {8..15,24..31}.
// ---------------------------------------------------------------------------
__global__ void k_pack(const float* __restrict__ W1, const float* __restrict__ W2,
                       const float* __restrict__ Wg,
                       __bf16* __restrict__ BGp,
                       __bf16* __restrict__ B1p,
                       __bf16* __restrict__ B2p) {
    int idx = blockIdx.x * 256 + threadIdx.x;
    __bf16* dst;
    int loc;
    int region;
    if (idx < BG_ELEMS)                            { region = 0; loc = idx;                        dst = BGp + loc; }
    else if (idx < BG_ELEMS + B1_ELEMS)            { region = 1; loc = idx - BG_ELEMS;             dst = B1p + loc; }
    else if (idx < BG_ELEMS + B1_ELEMS + B2_ELEMS) { region = 2; loc = idx - BG_ELEMS - B1_ELEMS;  dst = B2p + loc; }
    else return;

    int i      = loc & 15;
    int lane   = (loc >> 4) & 31;
    int tile   = loc >> 9;
    int lcol   = lane & 15;
    int klocal = (i < 8 ? i : i + 8) + ((lane >> 4) << 3);

    float val = 0.0f;
    if (region == 0) {                // gates: Wg (37,1536,4) -> N'=t*4+e, padded to 160
        int kt = tile / NGT, nt = tile % NGT;
        int np = nt * 16 + lcol;
        int k  = kt * 32 + klocal;
        if (np < NGCOL) val = Wg[(size_t)(np >> 2) * (IN_F * NEXP) + (size_t)k * NEXP + (np & 3)];
    } else if (region == 1) {         // experts L1: W1 (4,1536,64)
        int e  = tile / (KT * 4);
        int r  = tile % (KT * 4);
        int kt = r >> 2, nt = r & 3;
        int n  = nt * 16 + lcol;
        int k  = kt * 32 + klocal;
        val = W1[(size_t)e * (IN_F * HID) + (size_t)k * HID + n];
    } else {                          // experts L2: W2 (4,64,64)
        int e  = tile >> 3;
        int kt = (tile >> 2) & 1;
        int nt = tile & 3;
        int n  = nt * 16 + lcol;
        int k  = kt * 32 + klocal;
        val = W2[(size_t)e * (HID * HID) + (size_t)k * HID + n];
    }
    *dst = (__bf16)val;
}

// ---------------------------------------------------------------------------
// Kernel 1: gate logits GEMM (x @ Wg_flat, N=148 padded 160) + bias + softmax(E=4).
// Block = 128 threads = 4 waves, each wave owns a 16-row M tile.
// All 10 B fragments are preloaded per k-tile so the loads issue as one clause
// with multiple outstanding loadcnt, draining into 10 back-to-back WMMAs.
// ---------------------------------------------------------------------------
__global__ __launch_bounds__(128) void k_gates(const float* __restrict__ x,
                                               const float* __restrict__ bg,
                                               const __bf16* __restrict__ BGp,
                                               float* __restrict__ gates) {
    int tid  = threadIdx.x;
    int w    = tid >> 5;
    int lane = tid & 31;
    int half = lane >> 4;
    int lcol = lane & 15;
    int r0   = blockIdx.x * 64 + w * 16;

    const float* xrow = x + (size_t)(r0 + lcol) * IN_F;

    v8f gc[NGT];
#pragma unroll
    for (int i = 0; i < NGT; ++i) gc[i] = vzero8();

    for (int kt = 0; kt < KT; ++kt) {
        int pkt = (kt + 4 < KT) ? kt + 4 : KT - 1;
        __builtin_prefetch(xrow + pkt * 32 + half * 8, 0, 3);

        const __bf16* kb = BGp + (size_t)kt * NGT * 512;
        v16bf b[NGT];
#pragma unroll
        for (int nt = 0; nt < NGT; ++nt) b[nt] = load_b_frag(kb + nt * 512, lane);

        v16bf a = load_a_from_x(xrow, kt * 32 + half * 8);
#pragma unroll
        for (int nt = 0; nt < NGT; ++nt) gc[nt] = wmma_bf16(a, b[nt], gc[nt]);
    }

    // epilogue: bias + softmax over E=4 (4 consecutive lanes = one task)
#pragma unroll
    for (int nt = 0; nt < NGT; ++nt) {
        int  np    = nt * 16 + lcol;
        bool valid = (np < NGCOL);
        float bias = valid ? bg[np] : 0.0f;
#pragma unroll
        for (int j = 0; j < 8; ++j) {
            float v = gc[nt][j] + bias;
            float m = v;
            m = fmaxf(m, __shfl_xor(m, 1));
            m = fmaxf(m, __shfl_xor(m, 2));
            float ex = __expf(v - m);
            float s = ex;
            s += __shfl_xor(s, 1);
            s += __shfl_xor(s, 2);
            float g = ex / s;
            if (valid) gates[(size_t)(r0 + j + half * 8) * NGCOL + np] = g;
        }
    }
}

// ---------------------------------------------------------------------------
// Kernel 2: h = relu(x@W1+b1)  ->  expert_out = h@W2+b2  ->  out = gates @ expert_out.
// Block = 128 threads = 4 waves, wave owns 16 rows.  B fragments for expert e+1
// are loaded while expert e's 4 WMMAs execute (2-deep pipeline: 64 B-VGPRs).
// h round-trips through LDS (bf16) to convert C-layout -> A-fragments for GEMM2.
// Dynamic LDS: 4 waves * 16 rows * 256 cols * 2B = 32 KB.
// ---------------------------------------------------------------------------
__global__ __launch_bounds__(128) void k_experts(const float* __restrict__ x,
                                                 const float* __restrict__ b1,
                                                 const float* __restrict__ b2,
                                                 const __bf16* __restrict__ B1p,
                                                 const __bf16* __restrict__ B2p,
                                                 const float* __restrict__ gates,
                                                 float* __restrict__ out) {
    extern __shared__ __bf16 hls[];   // [4][16][256] bf16
    int tid  = threadIdx.x;
    int w    = tid >> 5;
    int lane = tid & 31;
    int half = lane >> 4;
    int lcol = lane & 15;
    int r0   = blockIdx.x * 64 + w * 16;

    const float* xrow = x + (size_t)(r0 + lcol) * IN_F;

    // ---- GEMM 1: h accumulators, 4 experts x 4 n-tiles ----
    v8f h[16];
#pragma unroll
    for (int i = 0; i < 16; ++i) h[i] = vzero8();

    for (int kt = 0; kt < KT; ++kt) {
        int pkt = (kt + 4 < KT) ? kt + 4 : KT - 1;
        __builtin_prefetch(xrow + pkt * 32 + half * 8, 0, 3);

        v16bf a = load_a_from_x(xrow, kt * 32 + half * 8);

        // 2-deep pipeline over experts: load group e+1 while computing group e
        v16bf bA[4], bB[4];
#pragma unroll
        for (int nt = 0; nt < 4; ++nt)
            bA[nt] = load_b_frag(B1p + (size_t)((0 * KT + kt) * 4 + nt) * 512, lane);

#pragma unroll
        for (int e = 0; e < NEXP; ++e) {
            v16bf (&cur)[4] = (e & 1) ? bB : bA;
            v16bf (&nxt)[4] = (e & 1) ? bA : bB;
            if (e < NEXP - 1) {
#pragma unroll
                for (int nt = 0; nt < 4; ++nt)
                    nxt[nt] = load_b_frag(B1p + (size_t)(((e + 1) * KT + kt) * 4 + nt) * 512, lane);
            }
#pragma unroll
            for (int nt = 0; nt < 4; ++nt)
                h[e * 4 + nt] = wmma_bf16(a, cur[nt], h[e * 4 + nt]);
        }
    }

    // ---- bias + ReLU, spill h to LDS as bf16 (C-layout -> row-major) ----
    __bf16* myh = hls + w * 16 * 256;
#pragma unroll
    for (int e = 0; e < NEXP; ++e) {
#pragma unroll
        for (int nt = 0; nt < 4; ++nt) {
            int n = nt * 16 + lcol;
            float bias = b1[e * HID + n];
#pragma unroll
            for (int j = 0; j < 8; ++j) {
                float v = fmaxf(h[e * 4 + nt][j] + bias, 0.0f);
                myh[(j + half * 8) * 256 + e * HID + n] = (__bf16)v;
            }
        }
    }
    __syncthreads();

    // ---- GEMM 2: expert_out = h @ W2 (K=64 per expert -> 2 k-tiles) ----
    v8f eo[16];
#pragma unroll
    for (int i = 0; i < 16; ++i) eo[i] = vzero8();

#pragma unroll
    for (int e = 0; e < NEXP; ++e) {
#pragma unroll
        for (int kt2 = 0; kt2 < 2; ++kt2) {
            const __bf16* ap = myh + lcol * 256 + e * HID + kt2 * 32 + half * 8;
            uint4 lo = *(const uint4*)(ap);        // K {0..7}+off
            uint4 hi = *(const uint4*)(ap + 16);   // K {16..23}+off
            u32x8 u;
            u[0] = lo.x; u[1] = lo.y; u[2] = lo.z; u[3] = lo.w;
            u[4] = hi.x; u[5] = hi.y; u[6] = hi.z; u[7] = hi.w;
            v16bf a = __builtin_bit_cast(v16bf, u);

            v16bf b[4];
#pragma unroll
            for (int nt = 0; nt < 4; ++nt)
                b[nt] = load_b_frag(B2p + (size_t)((e * 2 + kt2) * 4 + nt) * 512, lane);
#pragma unroll
            for (int nt = 0; nt < 4; ++nt)
                eo[e * 4 + nt] = wmma_bf16(a, b[nt], eo[e * 4 + nt]);
        }
    }

    // ---- bias b2 ----
#pragma unroll
    for (int e = 0; e < NEXP; ++e) {
#pragma unroll
        for (int nt = 0; nt < 4; ++nt) {
            float bb = b2[e * HID + nt * 16 + lcol];
#pragma unroll
            for (int j = 0; j < 8; ++j) eo[e * 4 + nt][j] += bb;
        }
    }

    // ---- combine: out[r][t][n] = sum_e gates[r][t][e] * eo[e][r][n] ----
    const float* grow = gates + (size_t)r0 * NGCOL;
    for (int t = 0; t < NTASK; ++t) {
#pragma unroll
        for (int j = 0; j < 8; ++j) {
            int row = j + half * 8;
            float4 g = *(const float4*)(grow + (size_t)row * NGCOL + t * 4);
#pragma unroll
            for (int nt = 0; nt < 4; ++nt) {
                float acc = g.x * eo[nt][j]      + g.y * eo[4 + nt][j]
                          + g.z * eo[8 + nt][j]  + g.w * eo[12 + nt][j];
                out[((size_t)(r0 + row) * NTASK + t) * HID + nt * 16 + lcol] = acc;
            }
        }
    }
}

// ---------------------------------------------------------------------------
// Host launch
// ---------------------------------------------------------------------------
extern "C" void kernel_launch(void* const* d_in, const int* in_sizes, int n_in,
                              void* d_out, int out_size, void* d_ws, size_t ws_size,
                              hipStream_t stream) {
    const float* x  = (const float*)d_in[0];
    const float* W1 = (const float*)d_in[1];
    const float* b1 = (const float*)d_in[2];
    const float* W2 = (const float*)d_in[3];
    const float* b2 = (const float*)d_in[4];
    const float* Wg = (const float*)d_in[5];
    const float* bg = (const float*)d_in[6];
    float* out = (float*)d_out;

    char* ws = (char*)d_ws;
    float*   gates = (float*)ws;
    __bf16*  BGp   = (__bf16*)(ws + OFF_BG);
    __bf16*  B1p   = (__bf16*)(ws + OFF_B1);
    __bf16*  B2p   = (__bf16*)(ws + OFF_B2);

    // 0) pack weights -> bf16 WMMA B-fragments (tiny, L2-resident afterwards)
    int packN = BG_ELEMS + B1_ELEMS + B2_ELEMS;
    k_pack<<<(packN + 255) / 256, 256, 0, stream>>>(W1, W2, Wg, BGp, B1p, B2p);

    // 1) gate GEMM + softmax -> gates workspace
    k_gates<<<BROWS / 64, 128, 0, stream>>>(x, bg, BGp, gates);

    // 2) experts (two GEMMs) + weighted combine -> out
    k_experts<<<BROWS / 64, 128, 32768, stream>>>(x, b1, b2, B1p, B2p, gates, out);
}
